// TemporalMamba_31576599560859
// MI455X (gfx1250) — matrix operand order repaired and checked by reading, not compiled, not run on hardware
//
#include <hip/hip_runtime.h>
#include <hip/hip_bf16.h>
#include <math.h>

// ---------------------------------------------------------------------------
// Problem constants (match reference)
// ---------------------------------------------------------------------------
#define BB      4
#define LL      1024
#define DM      512
#define DSTATE  16
#define NLAYERS 4
#define ED      1024            // EXPAND * D_MODEL
#define DTRANK  32
#define DCONV   4
#define NOUT_T  20              // N_FUT * N_OUT
#define MTOT    (BB * LL)       // 4096 rows

typedef __attribute__((ext_vector_type(16))) __bf16        v16bf;
typedef __attribute__((ext_vector_type(8)))  float         v8f;
typedef __attribute__((ext_vector_type(8)))  unsigned int  v8u;

__device__ __forceinline__ unsigned bfbits(float f) {
    unsigned u = __builtin_bit_cast(unsigned, f);
    return (u + 0x7FFFu + ((u >> 16) & 1u)) >> 16;         // RNE -> bf16 bits
}
__device__ __forceinline__ unsigned pack2bf(float lo, float hi) {
    return bfbits(lo) | (bfbits(hi) << 16);                // elem0 in [15:0]
}
__device__ __forceinline__ float silu_f(float x) {
    return x / (1.0f + __expf(-x));
}
__device__ __forceinline__ float softplus_f(float x) {
    return (x > 20.0f) ? x : __logf(1.0f + __expf(x));
}

// ---------------------------------------------------------------------------
// Generic WMMA GEMM:  C[M,N] = A[M,K] * W[N,K]^T   (all row-major f32)
// block = 128 threads (4 waves), block tile 64x64, wave tile 32x32, K step 32.
// Tiles staged in LDS as packed 2xbf16 dwords; emits v_wmma_f32_16x16x32_bf16.
// Requires: M%64==0, N%64==0, K%32==0, lda/ldw even (true for all call sites).
// ---------------------------------------------------------------------------
#define TM   64
#define TN   64
#define TK   32
#define WPR  (TK / 2)        // 16 packed words per row
#define LDW  (WPR + 1)       // +1 dword pad -> 17-dword row stride

__global__ __launch_bounds__(128)
void gemm_wmma_bf16(const float* __restrict__ A, int lda,
                    const float* __restrict__ W, int ldw,
                    float* __restrict__ C, int ldc,
                    int M, int N, int K)
{
    __shared__ unsigned As[TM][LDW];
    __shared__ unsigned Ws[TN][LDW];

    const int tid   = threadIdx.x;
    const int wave  = tid >> 5;          // 0..3
    const int lane  = tid & 31;
    const int half  = lane >> 4;         // 0/1
    const int l16   = lane & 15;
    const int wm    = (wave >> 1) * 32;  // wave row offset in block tile
    const int wn    = (wave & 1) * 32;   // wave col offset in block tile
    const int tileM = blockIdx.y * TM;
    const int tileN = blockIdx.x * TN;

    v8f acc[2][2] = {};

    for (int k0 = 0; k0 < K; k0 += TK) {
        // ---- stage tiles: float2 global load -> packed 2xbf16 -> ds_store_b32
        // 64 rows x 16 words = 1024 words per matrix; 128 threads -> 8 iters.
        for (int t = tid; t < TM * WPR; t += 128) {
            int r = t >> 4, c = t & (WPR - 1);
            const float2 a2 = *(const float2*)&A[(size_t)(tileM + r) * lda + k0 + 2 * c];
            const float2 w2 = *(const float2*)&W[(size_t)(tileN + r) * ldw + k0 + 2 * c];
            As[r][c] = pack2bf(a2.x, a2.y);
            Ws[r][c] = pack2bf(w2.x, w2.y);
        }
        if (k0 + TK < K) {   // hint next K tiles into cache (global_prefetch_b8)
            __builtin_prefetch(&A[(size_t)(tileM + (tid >> 1)) * lda + k0 + TK], 0, 0);
            __builtin_prefetch(&W[(size_t)(tileN + (tid >> 1)) * ldw + k0 + TK], 0, 0);
        }
        __syncthreads();

        // ---- gather fragments (packed dwords) per CDNA5 wave32 WMMA layouts
        //   A 16x32 bf16: elem i -> K = (i<8?0:16) + 8*half + (i&7)
        //   B 32x16 bf16: elem i -> K = (i<8?0:16) + 8*half + (i&7)
        v8u aw[2], bw[2];
#pragma unroll
        for (int s = 0; s < 2; ++s) {
            const int ar = wm + s * 16 + l16;
            const int br = wn + s * 16 + l16;
#pragma unroll
            for (int j = 0; j < 8; ++j) {
                int wcol = (j < 4) ? (half * 4 + j) : (8 + half * 4 + (j - 4));
                aw[s][j] = As[ar][wcol];
            }
#pragma unroll
            for (int j = 0; j < 8; ++j)
                bw[s][j] = Ws[br][(j < 4) ? (half * 4 + j) : (8 + half * 4 + (j - 4))];
        }
        v16bf afr[2], bfr[2];
#pragma unroll
        for (int s = 0; s < 2; ++s) {
            afr[s] = __builtin_bit_cast(v16bf, aw[s]);
            bfr[s] = __builtin_bit_cast(v16bf, bw[s]);
        }

        // ---- 2x2 WMMA
#pragma unroll
        for (int i = 0; i < 2; ++i)
#pragma unroll
            for (int j = 0; j < 2; ++j)
                acc[i][j] = __builtin_amdgcn_wmma_f32_16x16x32_bf16(
                    false, afr[i], false, bfr[j], (short)0, acc[i][j],
                    false, false);
        __syncthreads();
    }

    // ---- store: C/D frag layout: M = r + 8*half, N = lane&15
#pragma unroll
    for (int i = 0; i < 2; ++i)
#pragma unroll
        for (int j = 0; j < 2; ++j)
#pragma unroll
            for (int r = 0; r < 8; ++r) {
                int m = tileM + wm + i * 16 + r + 8 * half;
                int n = tileN + wn + j * 16 + l16;
                C[(size_t)m * ldc + n] = acc[i][j][r];
            }
}

// ---------------------------------------------------------------------------
// Input projection: x[bt,d] = sum_i seq[bt,i]*Wi[d,i] + bi[d]   (N_IN = 3)
// ---------------------------------------------------------------------------
__global__ void k_proj_in(const float* __restrict__ seq,
                          const float* __restrict__ Wi,
                          const float* __restrict__ bi,
                          float* __restrict__ x)
{
    int idx = blockIdx.x * blockDim.x + threadIdx.x;
    if (idx >= MTOT * DM) return;
    int d  = idx & (DM - 1);
    int bt = idx / DM;
    const float* s = seq + (size_t)bt * 3;
    const float* w = Wi  + (size_t)d * 3;
    x[idx] = s[0] * w[0] + s[1] * w[1] + s[2] * w[2] + bi[d];
}

// ---------------------------------------------------------------------------
// Depthwise causal conv (k=4, left pad 3) + bias + SiLU.
// Reads xb = xz[:, :ED] (row stride 2*ED), writes xbc[MTOT, ED].
// ---------------------------------------------------------------------------
__global__ void k_conv_silu(const float* __restrict__ xz,
                            const float* __restrict__ cw,   // [ED,4]
                            const float* __restrict__ cb,   // [ED]
                            float* __restrict__ xbc)
{
    int idx = blockIdx.x * blockDim.x + threadIdx.x;
    if (idx >= MTOT * ED) return;
    int e  = idx & (ED - 1);
    int bt = idx / ED;
    int t  = bt & (LL - 1);
    int b  = bt / LL;
    const float* w = cw + (size_t)e * DCONV;
    float acc = cb[e];
#pragma unroll
    for (int j = 0; j < DCONV; ++j) {
        int tt = t - (DCONV - 1) + j;
        if (tt >= 0)
            acc += w[j] * xz[((size_t)(b * LL + tt)) * (2 * ED) + e];
    }
    xbc[idx] = silu_f(acc);
}

// ---------------------------------------------------------------------------
// delta = softplus(dtraw + bdt[e])   (in place)
// ---------------------------------------------------------------------------
__global__ void k_softplus(float* __restrict__ dt,
                           const float* __restrict__ bdt)
{
    int idx = blockIdx.x * blockDim.x + threadIdx.x;
    if (idx >= MTOT * ED) return;
    int e = idx & (ED - 1);
    dt[idx] = softplus_f(dt[idx] + bdt[e]);
}

// ---------------------------------------------------------------------------
// Selective scan: one thread per (b,e) chain, 16 states in registers.
//   h_n <- exp(dt*A_n)*h_n + dt*xb*B_n ;  y = sum_n h_n*C_n + D*xb ; y*=silu(z)
// B_n, C_n live in dbc[:, 32:48] / dbc[:, 48:64]; z in xz[:, ED:2ED].
// ---------------------------------------------------------------------------
__global__ __launch_bounds__(256)
void k_scan(const float* __restrict__ xbc,
            const float* __restrict__ dt,
            const float* __restrict__ dbc,   // [MTOT, 64]
            const float* __restrict__ xz,    // for z gate
            const float* __restrict__ Alog,  // [ED, 16]
            const float* __restrict__ Dp,    // [ED]
            float* __restrict__ y)
{
    int tid = blockIdx.x * blockDim.x + threadIdx.x;
    if (tid >= BB * ED) return;
    int e = tid & (ED - 1);
    int b = tid / ED;

    float An[DSTATE], h[DSTATE];
#pragma unroll
    for (int n = 0; n < DSTATE; ++n) {
        An[n] = -__expf(Alog[(size_t)e * DSTATE + n]);
        h[n]  = 0.0f;
    }
    const float Dv = Dp[e];

    for (int t = 0; t < LL; ++t) {
        size_t row = (size_t)(b * LL + t);
        float xbv = xbc[row * ED + e];
        float dtv = dt [row * ED + e];
        float zv  = xz [row * (2 * ED) + ED + e];
        const float* bc = dbc + row * 64;
        float du  = dtv * xbv;
        float acc = 0.0f;
#pragma unroll
        for (int n = 0; n < DSTATE; ++n) {
            float dA = __expf(dtv * An[n]);
            h[n] = dA * h[n] + du * bc[DTRANK + n];          // B
            acc += h[n] * bc[DTRANK + DSTATE + n];           // C
        }
        y[row * ED + e] = (acc + Dv * xbv) * silu_f(zv);
    }
}

// ---------------------------------------------------------------------------
// Final layernorm (last timestep) + head: out[b, 0:20]
// one block per batch row, 256 threads
// ---------------------------------------------------------------------------
__global__ __launch_bounds__(256)
void k_final(const float* __restrict__ x,
             const float* __restrict__ ln_g, const float* __restrict__ ln_b,
             const float* __restrict__ Wo,   const float* __restrict__ bo,
             float* __restrict__ out)
{
    __shared__ float red[256];
    __shared__ float xn[DM];
    const int b   = blockIdx.x;
    const int tid = threadIdx.x;
    const float* row = x + ((size_t)(b * LL + (LL - 1))) * DM;

    float a0 = row[tid], a1 = row[tid + 256];
    red[tid] = a0 + a1;
    __syncthreads();
    for (int o = 128; o > 0; o >>= 1) {
        if (tid < o) red[tid] += red[tid + o];
        __syncthreads();
    }
    float mu = red[0] * (1.0f / DM);
    __syncthreads();

    float d0 = a0 - mu, d1 = a1 - mu;
    red[tid] = d0 * d0 + d1 * d1;
    __syncthreads();
    for (int o = 128; o > 0; o >>= 1) {
        if (tid < o) red[tid] += red[tid + o];
        __syncthreads();
    }
    float rstd = rsqrtf(red[0] * (1.0f / DM) + 1e-5f);
    __syncthreads();

    xn[tid]       = d0 * rstd * ln_g[tid]       + ln_b[tid];
    xn[tid + 256] = d1 * rstd * ln_g[tid + 256] + ln_b[tid + 256];
    __syncthreads();

    if (tid < NOUT_T) {
        const float* w = Wo + (size_t)tid * DM;
        float acc = bo[tid];
        for (int d = 0; d < DM; ++d) acc += w[d] * xn[d];
        out[b * NOUT_T + tid] = acc;
    }
}

// ---------------------------------------------------------------------------
// Host-side orchestration
// ---------------------------------------------------------------------------
extern "C" void kernel_launch(void* const* d_in, const int* in_sizes, int n_in,
                              void* d_out, int out_size, void* d_ws, size_t ws_size,
                              hipStream_t stream)
{
    const float* seq   = (const float*)d_in[0];
    const float* Wi    = (const float*)d_in[1];
    const float* bi    = (const float*)d_in[2];
    const float* Wo    = (const float*)d_in[3];
    const float* bo    = (const float*)d_in[4];
    const float* ln_g  = (const float*)d_in[5];
    const float* ln_b  = (const float*)d_in[6];
    const float* lWin  = (const float*)d_in[7];   // [4, 2048, 512]
    const float* lcw   = (const float*)d_in[8];   // [4, 1024, 1, 4]
    const float* lcb   = (const float*)d_in[9];   // [4, 1024]
    const float* lWx   = (const float*)d_in[10];  // [4, 64, 1024]
    const float* lWdt  = (const float*)d_in[11];  // [4, 1024, 32]
    const float* lbdt  = (const float*)d_in[12];  // [4, 1024]
    const float* lAlog = (const float*)d_in[13];  // [4, 1024, 16]
    const float* lD    = (const float*)d_in[14];  // [4, 1024]
    const float* lWout = (const float*)d_in[15];  // [4, 512, 1024]
    float* out = (float*)d_out;

    // workspace layout (floats)
    float* ws  = (float*)d_ws;
    float* x   = ws;                              // [4096, 512]
    float* xz  = x   + (size_t)MTOT * DM;         // [4096, 2048]
    float* xbc = xz  + (size_t)MTOT * 2 * ED;     // [4096, 1024]
    float* dbc = xbc + (size_t)MTOT * ED;         // [4096, 64]
    float* dt  = dbc + (size_t)MTOT * 64;         // [4096, 1024]
    float* y   = dt  + (size_t)MTOT * ED;         // [4096, 1024]

    const int EW = 256;

    // input projection
    k_proj_in<<<(MTOT * DM + EW - 1) / EW, EW, 0, stream>>>(seq, Wi, bi, x);

    for (int l = 0; l < NLAYERS; ++l) {
        const float* Win  = lWin  + (size_t)l * 2 * ED * DM;
        const float* cw   = lcw   + (size_t)l * ED * DCONV;
        const float* cb   = lcb   + (size_t)l * ED;
        const float* Wx   = lWx   + (size_t)l * (DTRANK + 2 * DSTATE) * ED;
        const float* Wdt  = lWdt  + (size_t)l * ED * DTRANK;
        const float* bdt  = lbdt  + (size_t)l * ED;
        const float* Alog = lAlog + (size_t)l * ED * DSTATE;
        const float* Dp   = lD    + (size_t)l * ED;
        const float* Wout = lWout + (size_t)l * DM * ED;

        // xz = x @ Win^T   : M=4096 N=2048 K=512
        gemm_wmma_bf16<<<dim3(2 * ED / TN, MTOT / TM), 128, 0, stream>>>(
            x, DM, Win, DM, xz, 2 * ED, MTOT, 2 * ED, DM);

        // depthwise causal conv + SiLU -> xbc
        k_conv_silu<<<(MTOT * ED + EW - 1) / EW, EW, 0, stream>>>(xz, cw, cb, xbc);

        // dbc = xbc @ Wx^T : M=4096 N=64 K=1024
        gemm_wmma_bf16<<<dim3(64 / TN, MTOT / TM), 128, 0, stream>>>(
            xbc, ED, Wx, ED, dbc, 64, MTOT, 64, ED);

        // dtraw = dbc[:, :32] @ Wdt^T : M=4096 N=1024 K=32
        gemm_wmma_bf16<<<dim3(ED / TN, MTOT / TM), 128, 0, stream>>>(
            dbc, 64, Wdt, DTRANK, dt, ED, MTOT, ED, DTRANK);

        // delta = softplus(dtraw + bdt)
        k_softplus<<<(MTOT * ED + EW - 1) / EW, EW, 0, stream>>>(dt, bdt);

        // selective scan + D skip + z gating -> y
        k_scan<<<(BB * ED + EW - 1) / EW, EW, 0, stream>>>(
            xbc, dt, dbc, xz, Alog, Dp, y);

        // x = y @ Wout^T : M=4096 N=512 K=1024   (overwrites layer input)
        gemm_wmma_bf16<<<dim3(DM / TN, MTOT / TM), 128, 0, stream>>>(
            y, ED, Wout, ED, x, DM, MTOT, DM, ED);
    }

    // layernorm on last timestep + output head
    k_final<<<BB, 256, 0, stream>>>(x, ln_g, ln_b, Wo, bo, out);
}
